// MEF_Loss_84129819394243
// MI455X (gfx1250) — matrix operand order repaired and checked by reading, not compiled, not run on hardware
//
#include <hip/hip_runtime.h>
#include <hip/hip_bf16.h>
#include <math.h>

typedef __attribute__((ext_vector_type(16))) _Float16 v16h;
typedef __attribute__((ext_vector_type(8)))  _Float16 v8h;
typedef __attribute__((ext_vector_type(8)))  float    v8f;

#define IMG_W 512
#define IMG_H 512
#define HWN   (512*512)
#define NBATCH 4

// ws layout (floats):
//   p1 partials: 12288 blocks * 3            -> [0, 36864)
//   p2 partials: 12288 blocks * 1 (cs sum)   -> [36864, 49152)
//   p3 partials: 8192 blocks * 2 (num,den)   -> [49152, 65536)
#define P1_BLOCKS 12288
#define P2_BLOCKS 12288
#define P3_BLOCKS 8192
#define P2_OFF (P1_BLOCKS*3)
#define P3_OFF (P2_OFF + P2_BLOCKS)

__device__ __forceinline__ float clip01(float v) { return fminf(fmaxf(v, 0.f), 1.f); }

// 11-tap Gaussian (sigma=1.5), normalized: g[t] = exp(-(t-5)^2/4.5)/Z, Z=3.75923278
__device__ __forceinline__ float gwin(int t) {
  float d = (float)(t - 5);
  return expf(-(d * d) * (1.0f / 4.5f)) * 0.26601179f;
}

__device__ __forceinline__ void ycc(float r, float g, float b,
                                    float& y, float& cr, float& cb) {
  float yy = 0.299f * r + 0.587f * g + 0.114f * b;
  cr = clip01((r - yy) * 0.713f + 0.5f);
  cb = clip01((b - yy) * 0.564f + 0.5f);
  y = clip01(yy);
}

__device__ __forceinline__ float block_reduce_256(float v, float* sm) {
  int t = threadIdx.x;
  sm[t] = v; __syncthreads();
  for (int s = 128; s > 0; s >>= 1) {
    if (t < s) sm[t] += sm[t + s];
    __syncthreads();
  }
  float r = sm[0]; __syncthreads();
  return r;
}

// ---------------------------------------------------------------------------
// Pass 1: elementwise losses (l_int, l_grad, l_color), per 16x16 tile.
// blocks: b(4) * frame(3) * tiles(1024) = 12288, 256 threads.
// ---------------------------------------------------------------------------
__global__ __launch_bounds__(256) void pass1_elemwise(
    const float* __restrict__ f, const float* __restrict__ a,
    const float* __restrict__ b, float* __restrict__ part) {
  __shared__ float sY[3][18 * 18];
  __shared__ float sred[256];
  int bid  = blockIdx.x;
  int tile = bid & 1023;
  int fi   = (bid >> 10) % 3;
  int bb   = bid / 3072;
  int tx = tile & 31, ty = tile >> 5;
  int tid = threadIdx.x;

  const float* base0 = f + (size_t)((bb * 3 + fi) * 3) * HWN;
  const float* base1 = a + (size_t)((bb * 5 + fi + 1) * 3) * HWN;
  const float* base2 = b + (size_t)((bb * 5 + fi + 1) * 3) * HWN;
  const float* bases[3] = { base0, base1, base2 };

  // Stage Y (zero-padded halo of 1) for sobel.
  for (int img = 0; img < 3; ++img) {
    const float* p = bases[img];
    for (int idx = tid; idx < 18 * 18; idx += 256) {
      int row = idx / 18, col = idx % 18;
      int gy = ty * 16 - 1 + row, gx = tx * 16 - 1 + col;
      float y = 0.f;
      if (gy >= 0 && gy < IMG_H && gx >= 0 && gx < IMG_W) {
        size_t o = (size_t)gy * IMG_W + gx;
        __builtin_prefetch(p + o + IMG_W, 0, 1);   // global_prefetch_b8
        float r = p[o], g = p[o + HWN], bl = p[o + 2 * HWN];
        y = clip01(0.299f * r + 0.587f * g + 0.114f * bl);
      }
      sY[img][idx] = y;
    }
  }
  __syncthreads();

  int lx = tid & 15, ly = tid >> 4;
  int gx = tx * 16 + lx, gy = ty * 16 + ly;

  float gmag[3];
  #pragma unroll
  for (int img = 0; img < 3; ++img) {
    const float* s = sY[img];
    int cpos = (ly + 1) * 18 + (lx + 1);
    float t00 = s[cpos - 19], t01 = s[cpos - 18], t02 = s[cpos - 17];
    float t10 = s[cpos - 1],                      t12 = s[cpos + 1];
    float t20 = s[cpos + 17], t21 = s[cpos + 18], t22 = s[cpos + 19];
    float gxs = (t02 - t00) + 2.f * (t12 - t10) + (t22 - t20);
    float gys = (t20 - t00) + 2.f * (t21 - t01) + (t22 - t02);
    gmag[img] = fabsf(gxs) + fabsf(gys);
  }
  int cpos = (ly + 1) * 18 + (lx + 1);
  float fyY = sY[0][cpos], aY = sY[1][cpos], bY = sY[2][cpos];
  float v_int  = fabsf(fyY - 0.5f * (aY + bY));
  float v_grad = fabsf(gmag[0] - fmaxf(gmag[1], gmag[2]));

  // Color term: Cr/Cb of center pixels.
  size_t o = (size_t)gy * IMG_W + gx;
  float yd, fcr, fcb, acr, acb, bcr, bcb;
  ycc(base0[o], base0[o + HWN], base0[o + 2 * HWN], yd, fcr, fcb);
  ycc(base1[o], base1[o + HWN], base1[o + 2 * HWN], yd, acr, acb);
  ycc(base2[o], base2[o + HWN], base2[o + 2 * HWN], yd, bcr, bcb);
  float v_col = 0.f;
  {
    float wa = fabsf(acr - 0.5f), wb = fabsf(bcr - 0.5f);
    v_col += fabsf(fcr - (acr * wa + bcr * wb) / (wa + wb + 1e-8f));
    wa = fabsf(acb - 0.5f); wb = fabsf(bcb - 0.5f);
    v_col += fabsf(fcb - (acb * wa + bcb * wb) / (wa + wb + 1e-8f));
  }

  float si = block_reduce_256(v_int,  sred);
  float sg = block_reduce_256(v_grad, sred);
  float sc = block_reduce_256(v_col,  sred);
  if (tid == 0) {
    part[bid * 3 + 0] = si;
    part[bid * 3 + 1] = sg;
    part[bid * 3 + 2] = sc;
  }
}

// ---------------------------------------------------------------------------
// Pass 2: MEF-SSIM via WMMA separable 11x11 Gaussian conv.
// blocks: b(4) * frame(3) * tiles(1024) = 12288, 256 threads = 8 waves.
// Wave w owns conv field w; each field is plane[p0] * plane[p1] (or *1):
//   f0=X f1=X*X f2=Ya f3=Ya*Ya f4=Yb f5=Yb*Yb f6=X*Ya f7=X*Yb
// A-fragments are built branchlessly from 32-row zero-padded planes with
// contiguous v8f LDS loads (A-layout K indices form two 8-element runs).
// H->V handoff staged per-wave in column-major f16 LDS (b128 store/loads).
// ---------------------------------------------------------------------------
__global__ __launch_bounds__(256) void pass2_ssim(
    const float* __restrict__ f, const float* __restrict__ a,
    const float* __restrict__ b, float* __restrict__ part) {
  __shared__ float s_rgb[9][26 * 32];              // raw RGB (async-staged)
  __shared__ __align__(32) float s_base[3][32 * 32]; // X,Ya,Yb; rows 26..31 = 0
  __shared__ v8h   s_h[8][64];   // per-wave h-conv staging, col-major 32x16 f16
  __shared__ v8f   s_res[8][32]; // v-conv results, col-major 16x16 f32 per field
  __shared__ float s_red[256];

  int bid  = blockIdx.x;
  int tile = bid & 1023;
  int fi   = (bid >> 10) % 3;
  int bb   = bid / 3072;
  int tx = tile & 31, ty = tile >> 5;
  int gx0 = tx * 16 - 5, gy0 = ty * 16 - 5;  // 26x26 input window (pad 5)
  int tid  = threadIdx.x;
  int wave = tid >> 5;          // 0..7 == conv field id
  int lane = tid & 31;
  int laneLo = lane & 15, hi = lane >> 4;

  // Zero LDS (zero-pad for out-of-tile / out-of-image, incl. pad rows 26..31).
  for (int idx = tid; idx < 26 * 32; idx += 256)
    #pragma unroll
    for (int p = 0; p < 9; ++p) s_rgb[p][idx] = 0.f;
  for (int idx = tid; idx < 3 * 32 * 32; idx += 256) ((float*)s_base)[idx] = 0.f;
  __syncthreads();

  unsigned long long bases[9];
  #pragma unroll
  for (int ch = 0; ch < 3; ++ch) {
    bases[ch]     = (unsigned long long)(f + (size_t)((bb * 3 + fi) * 3 + ch) * HWN);
    bases[3 + ch] = (unsigned long long)(a + (size_t)((bb * 5 + fi + 1) * 3 + ch) * HWN);
    bases[6 + ch] = (unsigned long long)(b + (size_t)((bb * 5 + fi + 1) * 3 + ch) * HWN);
  }

  // Async global->LDS staging (EXEC mask drops out-of-range lanes -> stays 0).
  for (int idx = tid; idx < 26 * 32; idx += 256) {
    int row = idx >> 5, col = idx & 31;
    int gy = gy0 + row, gx = gx0 + col;
    if (gy >= 0 && gy < IMG_H && gx >= 0 && gx < IMG_W && col < 26) {
      unsigned voff = ((unsigned)gy * IMG_W + (unsigned)gx) * 4u;
      #pragma unroll
      for (int p = 0; p < 9; ++p) {
        unsigned lds = (unsigned)(unsigned long long)&s_rgb[p][idx];
        asm volatile("global_load_async_to_lds_b32 %0, %1, %2"
                     :: "v"(lds), "v"(voff), "s"(bases[p]) : "memory");
      }
    }
  }
  asm volatile("s_wait_asynccnt 0x0" ::: "memory");
  __syncthreads();

  // Constant banded-coefficient fragments (ISA 7.12.2 layouts):
  // A-matrix f16 16x32: M = lane&15; element e -> K = e + 8*hi + (e>=8 ? 8 : 0)
  // B-matrix f16 32x16: N = lane&15; element e -> K = e + 8*hi + (e>=8 ? 8 : 0)
  v16h bfrag_wh, afrag_wv;
  #pragma unroll
  for (int e = 0; e < 16; ++e) {
    int Ka = e + 8 * hi + ((e >= 8) ? 8 : 0);
    int Kb = e + 8 * hi + ((e >= 8) ? 8 : 0);
    int ta = Ka - laneLo, tb = Kb - laneLo;
    afrag_wv[e] = (_Float16)((ta >= 0 && ta < 11) ? gwin(ta) : 0.f);  // Wv[m][k]=g[k-m]
    bfrag_wh[e] = (_Float16)((tb >= 0 && tb < 11) ? gwin(tb) : 0.f);  // Wh[k][n]=g[k-n]
  }

  // Wave-uniform field -> source-plane mapping (branchless fragment build).
  int fld = wave;
  int p0 = (fld == 2 || fld == 3 || fld == 6) ? 1
         : (fld == 4 || fld == 5 || fld == 7) ? 2 : 0;
  int p1 = (fld == 6) ? 0 : (fld == 7) ? 0 : p0;   // f6=Ya*X, f7=Yb*X
  bool up = (fld == 1) || (fld == 3) || (fld == 5) || (fld == 6) || (fld == 7);
  const v8f* s0v = (const v8f*)&s_base[0][0] + p0 * 128;  // plane = 1024 f = 128 v8f
  const v8f* s1v = (const v8f*)&s_base[0][0] + p1 * 128;

  const float C2 = 0.0009f;  // 0.03^2
  float cs_local = 0.f;
  const float* rf = (const float*)s_res;

  #pragma unroll
  for (int c = 0; c < 3; ++c) {
    // Build zero-padded YCrCb field bases for channel c (rows 0..25 only).
    __syncthreads();
    for (int idx = tid; idx < 26 * 32; idx += 256) {
      int row = idx >> 5, col = idx & 31;
      int gy = gy0 + row, gx = gx0 + col;
      bool valid = (gy >= 0 && gy < IMG_H && gx >= 0 && gx < IMG_W && col < 26);
      #pragma unroll
      for (int img = 0; img < 3; ++img) {
        float val = 0.f;
        if (valid) {
          float r = s_rgb[img * 3 + 0][idx];
          float g = s_rgb[img * 3 + 1][idx];
          float bl = s_rgb[img * 3 + 2][idx];
          float y = 0.299f * r + 0.587f * g + 0.114f * bl;
          if (c == 0)      val = clip01(y);
          else if (c == 1) val = clip01((r - y) * 0.713f + 0.5f);
          else             val = clip01((bl - y) * 0.564f + 0.5f);
        }
        s_base[img][row * 32 + col] = val;
      }
    }
    __syncthreads();

    // --- this wave's field: horizontal conv, two 16-row groups ---
    #pragma unroll
    for (int rg = 0; rg < 2; ++rg) {
      int row = rg * 16 + laneLo;  // 0..31; rows >= 26 read zero padding
      v8f x0 = s0v[row * 4 + hi];      // K = 8*hi + 0..7
      v8f x1 = s0v[row * 4 + 2 + hi];  // K = 16 + 8*hi + 0..7
      v8f m0 = s1v[row * 4 + hi];
      v8f m1 = s1v[row * 4 + 2 + hi];
      v16h afrag;
      #pragma unroll
      for (int e = 0; e < 8; ++e) {
        float v0 = x0[e] * (up ? m0[e] : 1.f);
        float v1 = x1[e] * (up ? m1[e] : 1.f);
        afrag[e]     = (_Float16)v0;
        afrag[e + 8] = (_Float16)v1;
      }
      v8f cz = {0.f, 0.f, 0.f, 0.f, 0.f, 0.f, 0.f, 0.f};
      v8f d = __builtin_amdgcn_wmma_f32_16x16x32_f16(
          false, afrag, false, bfrag_wh, (short)0, cz, false, false);
      // D: M = r + 8*hi (+16*rg), N = lane&15. Column-major store -> one b128.
      v8h hv;
      #pragma unroll
      for (int r = 0; r < 8; ++r) hv[r] = (_Float16)d[r];
      s_h[wave][laneLo * 4 + rg * 2 + hi] = hv;
    }
    // Wave-private LDS: hardware keeps per-wave DS ops in order; just stop the
    // compiler from reordering the dependent loads above the stores.
    __builtin_amdgcn_wave_barrier();
    asm volatile("" ::: "memory");

    // --- vertical conv: A = Wv (banded const), B = h-conv (2 b128 loads) ---
    v8h blo = s_h[wave][laneLo * 4 + hi];
    v8h bhi = s_h[wave][laneLo * 4 + 2 + hi];
    v16h bfrag;
    #pragma unroll
    for (int e = 0; e < 8; ++e) { bfrag[e] = blo[e]; bfrag[e + 8] = bhi[e]; }
    v8f cz = {0.f, 0.f, 0.f, 0.f, 0.f, 0.f, 0.f, 0.f};
    v8f dv = __builtin_amdgcn_wmma_f32_16x16x32_f16(
        false, afrag_wv, false, bfrag, (short)0, cz, false, false);
    // Col-major f32 result: pixels (col=lane&15, rows 8*hi..8*hi+7) contiguous.
    s_res[fld][laneLo * 2 + hi] = dv;
    __syncthreads();

    // --- per-pixel cs term: each thread handles one of the 256 tile pixels ---
    {
      int p = tid;  // p = col*16 + row (col-major), coords irrelevant for the sum
      float muX = rf[0 * 256 + p], exx = rf[1 * 256 + p];
      float muA = rf[2 * 256 + p], eaa = rf[3 * 256 + p];
      float muB = rf[4 * 256 + p], ebb = rf[5 * 256 + p];
      float exa = rf[6 * 256 + p], exb = rf[7 * 256 + p];
      float sigX = exx - muX * muX;
      float sigA = eaa - muA * muA;
      float sigB = ebb - muB * muB;
      float sxa = exa - muX * muA;
      float sxb = exb - muX * muB;
      bool useA = (sigA >= sigB);  // argmax over K, first index wins on tie
      float sY  = useA ? sigA : sigB;
      float sXY = useA ? sxa : sxb;
      cs_local += (2.f * sXY + C2) / (sigX + sY + C2);
    }
    __syncthreads();  // before next channel overwrites s_base / s_res
  }

  float t = block_reduce_256(cs_local, s_red);
  if (tid == 0) part[bid] = t;
}

// ---------------------------------------------------------------------------
// Pass 3: temporal loss (bilinear flow warp + masked L1).
// blocks: b(4) * dir(2) * tiles(1024) = 8192, 256 threads.
// ---------------------------------------------------------------------------
__global__ __launch_bounds__(256) void pass3_temporal(
    const float* __restrict__ f, const float* __restrict__ flow_prev,
    const float* __restrict__ flow_next, float* __restrict__ part) {
  __shared__ float sred[256];
  int bid  = blockIdx.x;
  int tile = bid & 1023;
  int dir  = (bid >> 10) & 1;
  int bb   = bid >> 11;
  int tx = tile & 31, ty = tile >> 5;
  int tid = threadIdx.x;
  int lx = tid & 15, ly = tid >> 4;
  int gx = tx * 16 + lx, gy = ty * 16 + ly;

  const float* flow = dir ? flow_next : flow_prev;
  const float* img  = f + (size_t)((bb * 3 + (dir ? 2 : 0)) * 3) * HWN;
  const float* cur  = f + (size_t)((bb * 3 + 1) * 3) * HWN;

  size_t o = (size_t)gy * IMG_W + gx;
  float u = flow[(size_t)(bb * 2 + 0) * HWN + o];
  float v = flow[(size_t)(bb * 2 + 1) * HWN + o];
  float xf = (float)gx + u, yf = (float)gy + v;
  float x0 = floorf(xf), y0 = floorf(yf);
  float wx = xf - x0, wy = yf - y0;
  int xi0 = (int)x0, yi0 = (int)y0;
  int xa = xi0 < 0 ? 0 : (xi0 > IMG_W - 1 ? IMG_W - 1 : xi0);
  int xb2 = xi0 + 1 < 0 ? 0 : (xi0 + 1 > IMG_W - 1 ? IMG_W - 1 : xi0 + 1);
  int ya = yi0 < 0 ? 0 : (yi0 > IMG_H - 1 ? IMG_H - 1 : yi0);
  int yb2 = yi0 + 1 < 0 ? 0 : (yi0 + 1 > IMG_H - 1 ? IMG_H - 1 : yi0 + 1);

  float m = (fmaxf(fabsf(u), fabsf(v)) > 3.f) ? 0.f : 1.f;
  float d = 0.f;
  #pragma unroll
  for (int ch = 0; ch < 3; ++ch) {
    const float* ip = img + (size_t)ch * HWN;
    float g00 = ip[(size_t)ya * IMG_W + xa];
    float g01 = ip[(size_t)ya * IMG_W + xb2];
    float g10 = ip[(size_t)yb2 * IMG_W + xa];
    float g11 = ip[(size_t)yb2 * IMG_W + xb2];
    float r = (1.f - wy) * ((1.f - wx) * g00 + wx * g01) +
              wy * ((1.f - wx) * g10 + wx * g11);
    d += fabsf(cur[(size_t)ch * HWN + o] - r);
  }
  d *= (1.f / 3.f);

  float num = block_reduce_256(m * d, sred);
  float den = block_reduce_256(m, sred);
  if (tid == 0) {
    part[bid * 2 + 0] = num;
    part[bid * 2 + 1] = den;
  }
}

// ---------------------------------------------------------------------------
// Finalize: deterministic fixed-order reduction of all partials -> 6 outputs.
// ---------------------------------------------------------------------------
__global__ __launch_bounds__(256) void finalize_kernel(
    const float* __restrict__ ws, float* __restrict__ out) {
  __shared__ float sm[256];
  int t = threadIdx.x;
  const float* p1 = ws;
  const float* p2 = ws + P2_OFF;
  const float* p3 = ws + P3_OFF;

  float si = 0.f, sg = 0.f, sc = 0.f, ss = 0.f;
  for (int i = t; i < P1_BLOCKS; i += 256) {
    si += p1[i * 3 + 0];
    sg += p1[i * 3 + 1];
    sc += p1[i * 3 + 2];
  }
  for (int i = t; i < P2_BLOCKS; i += 256) ss += p2[i];
  si = block_reduce_256(si, sm);
  sg = block_reduce_256(sg, sm);
  sc = block_reduce_256(sc, sm);
  ss = block_reduce_256(ss, sm);

  float lt_sum = 0.f;
  for (int grp = 0; grp < 8; ++grp) {  // grp = b*2 + dir
    float n = 0.f, d = 0.f;
    for (int i = t; i < 1024; i += 256) {
      n += p3[(grp * 1024 + i) * 2 + 0];
      d += p3[(grp * 1024 + i) * 2 + 1];
    }
    n = block_reduce_256(n, sm);
    d = block_reduce_256(d, sm);
    lt_sum += n / (d + 1e-10f);
  }

  if (t == 0) {
    float li = si / (4.f * (float)HWN);        // mean over B*1*H*W, summed 3 frames
    float lg = sg / (4.f * (float)HWN);
    float lc = sc / (8.f * (float)HWN);        // mean over B*2*H*W
    float ls = 3.f - ss / (12.f * (float)HWN); // sum_i (1 - mean_i), mean over B*3*H*W
    float lt = lt_sum / (float)NBATCH;
    float total = (li + lc) + lg + ls + 0.1f * lt;
    out[0] = total; out[1] = li; out[2] = lg;
    out[3] = lc;    out[4] = ls; out[5] = lt;
  }
}

extern "C" void kernel_launch(void* const* d_in, const int* in_sizes, int n_in,
                              void* d_out, int out_size, void* d_ws, size_t ws_size,
                              hipStream_t stream) {
  const float* f  = (const float*)d_in[0];
  const float* a  = (const float*)d_in[1];
  const float* b  = (const float*)d_in[2];
  const float* fp = (const float*)d_in[3];
  const float* fn = (const float*)d_in[4];
  float* ws  = (float*)d_ws;
  float* out = (float*)d_out;

  pass1_elemwise<<<P1_BLOCKS, 256, 0, stream>>>(f, a, b, ws);
  pass2_ssim<<<P2_BLOCKS, 256, 0, stream>>>(f, a, b, ws + P2_OFF);
  pass3_temporal<<<P3_BLOCKS, 256, 0, stream>>>(f, fp, fn, ws + P3_OFF);
  finalize_kernel<<<1, 256, 0, stream>>>(ws, out);
}
